// SelfAttention_28097676051290
// MI455X (gfx1250) — matrix operand order repaired and checked by reading, not compiled, not run on hardware
//
#include <hip/hip_runtime.h>
#include <hip/hip_bf16.h>

// Self-attention (SAGAN-style), B=4 C=256 N=4096 CQ=32, column-normalized softmax.
// Flash-style: never materialize attn [B,N,N]; bf16 WMMA with fp32 accumulation.

#define BB 4
#define CC 256
#define NN 4096
#define CQ 32

#define AS1 __attribute__((address_space(1)))
#define AS3 __attribute__((address_space(3)))

typedef __bf16 bf16_t;
typedef __attribute__((ext_vector_type(8)))  bf16_t v8bf;
typedef __attribute__((ext_vector_type(16))) bf16_t v16bf;
typedef __attribute__((ext_vector_type(8)))  float  v8f;
typedef __attribute__((ext_vector_type(4)))  float  v4f;

static __device__ __forceinline__ v16bf cat16(v8bf a, v8bf b) {
  return __builtin_shufflevector(a, b, 0,1,2,3,4,5,6,7,8,9,10,11,12,13,14,15);
}

static __device__ __forceinline__ v8f wmma_bf16(v16bf a, v16bf b, v8f c) {
  // D = A(16x32) * B(32x16) + C, f32 accumulate
  return __builtin_amdgcn_wmma_f32_16x16x32_bf16(false, a, false, b, (short)0, c, false, false);
}

// A-fragment (16x32 bf16) from row-major [16][32] bf16 buffer.
// lane<16: row=lane,     K chunks at +0 and +16
// lane>=16: row=lane-16, K chunks at +8 and +24
static __device__ __forceinline__ v16bf load_a32(const bf16_t* base, int lane) {
  const bf16_t* p = base + (lane & 15) * 32 + ((lane >> 4) * 8);
  v8bf lo = *reinterpret_cast<const v8bf*>(p);
  v8bf hi = *reinterpret_cast<const v8bf*>(p + 16);
  return cat16(lo, hi);
}

// B-fragment (32x16 bf16): column j of B = contiguous 32-element row j of `base`.
// lane<16: col=lane, K chunks at +0 and +16; lane>=16: col=lane-16, K chunks at +8 and +24.
static __device__ __forceinline__ v16bf load_b_rows(const bf16_t* base, int stride, int lane) {
  const bf16_t* p = base + (size_t)(lane & 15) * stride + ((lane >> 4) * 8);
  v8bf lo = *reinterpret_cast<const v8bf*>(p);
  v8bf hi = *reinterpret_cast<const v8bf*>(p + 16);
  return cat16(lo, hi);
}

// Build a bf16 fragment from two 8-float chunks (f32 source -> v_cvt_pk_bf16).
static __device__ __forceinline__ v16bf frag_f32_2x8(const float* p0, const float* p1) {
  v4f a0 = *reinterpret_cast<const v4f*>(p0);
  v4f a1 = *reinterpret_cast<const v4f*>(p0 + 4);
  v4f a2 = *reinterpret_cast<const v4f*>(p1);
  v4f a3 = *reinterpret_cast<const v4f*>(p1 + 4);
  v16bf f;
#pragma unroll
  for (int r = 0; r < 4; ++r) {
    f[r]      = (bf16_t)a0[r];
    f[4 + r]  = (bf16_t)a1[r];
    f[8 + r]  = (bf16_t)a2[r];
    f[12 + r] = (bf16_t)a3[r];
  }
  return f;
}

// ---------------------------------------------------------------------------
// Kernel 1: fused QKV projection (bf16 WMMA, K=256 in 8 slabs of 32).
// Wave teaming: team = w>>1 owns otiles [team*5, team*5+5); half = w&1 owns
// n-tiles {half*2, half*2+1}. A-frags reused x2, B-frags reused x5.
//   qT[b][n][o] (bf16)   kT[b][n][o] (bf16)   vT[b][c][n] (bf16)
// ---------------------------------------------------------------------------
__global__ __launch_bounds__(256) void qkv_kernel(
    const float* __restrict__ x,
    const float* __restrict__ Wq, const float* __restrict__ bq,
    const float* __restrict__ Wk, const float* __restrict__ bk,
    const float* __restrict__ Wv, const float* __restrict__ bv,
    bf16_t* __restrict__ qT, bf16_t* __restrict__ kT, bf16_t* __restrict__ vT) {
  __shared__ __align__(16) char smem[256 * 72 * 2];          // 36 KB
  float*  xsf    = reinterpret_cast<float*>(smem);           // [64][36] f32 (x^T tile)
  bf16_t* vstage = reinterpret_cast<bf16_t*>(smem);          // [256][72] bf16 (reused)

  const int t     = threadIdx.x;
  const int lane  = t & 31;
  const int w     = t >> 5;
  const int team  = w >> 1;       // 0..3
  const int half  = w & 1;        // 0..1
  const int b     = blockIdx.y;
  const int n0    = blockIdx.x * 64;
  const int row_a = lane & 15;
  const int kb    = (lane >> 4) * 8;

  v8f acc[10];  // acc[ii*2+jj] : otile=team*5+ii, ntile=half*2+jj
#pragma unroll
  for (int i = 0; i < 10; ++i) acc[i] = v8f{0.f,0.f,0.f,0.f,0.f,0.f,0.f,0.f};

  for (int ks = 0; ks < 8; ++ks) {
    const int k0 = ks * 32;
    __syncthreads();
    // Load x[k0..k0+32)[n0..n0+64) transposed into LDS as f32 [n][k] (stride 36).
    {
      const float* xrow = x + ((size_t)b * CC + k0) * NN + n0;
#if __has_builtin(__builtin_amdgcn_global_load_async_to_lds_b32)
      // CDNA5 async global->LDS with per-lane LDS scatter address (free transpose)
#pragma unroll
      for (int i = 0; i < 8; ++i) {
        int e = i * 256 + t, kk = e >> 6, nn = e & 63;
        __builtin_amdgcn_global_load_async_to_lds_b32(
            (AS1 int*)(xrow + (size_t)kk * NN + nn),
            (AS3 int*)(xsf + nn * 36 + kk), 0, 0);
      }
      asm volatile("s_wait_asynccnt 0" ::: "memory");
#else
#pragma unroll
      for (int i = 0; i < 8; ++i) {
        int e = i * 256 + t, kk = e >> 6, nn = e & 63;
        xsf[nn * 36 + kk] = xrow[(size_t)kk * NN + nn];
      }
#endif
    }
    __syncthreads();

    // B-fragments for this wave's two n-tiles (f32 LDS -> bf16), reused over 5 otiles.
    v16bf bfr[2];
#pragma unroll
    for (int jj = 0; jj < 2; ++jj) {
      int nt = half * 2 + jj;
      const float* p = xsf + (nt * 16 + row_a) * 36 + ((lane >> 4) * 8);
      bfr[jj] = frag_f32_2x8(p, p + 16);
    }

#pragma unroll
    for (int ii = 0; ii < 5; ++ii) {
      const int ot = team * 5 + ii;
      const float* Wbase; int orow;
      if      (ot < 2) { Wbase = Wq; orow = ot * 16; }
      else if (ot < 4) { Wbase = Wk; orow = (ot - 2) * 16; }
      else             { Wbase = Wv; orow = (ot - 4) * 16; }
      const float* wp = Wbase + (size_t)(orow + row_a) * CC + k0 + kb;
      v16bf af = frag_f32_2x8(wp, wp + 16);
#pragma unroll
      for (int jj = 0; jj < 2; ++jj)
        acc[ii * 2 + jj] = wmma_bf16(af, bfr[jj], acc[ii * 2 + jj]);
    }
  }
  __syncthreads();  // xsf no longer needed; vstage reuses smem

  // Epilogue: bias, store q/k (coalesced b128), stage v through LDS.
#pragma unroll
  for (int ii = 0; ii < 5; ++ii) {
    const int ot = team * 5 + ii;
    const float* bias; int orow;
    if      (ot < 2) { bias = bq; orow = ot * 16; }
    else if (ot < 4) { bias = bk; orow = (ot - 2) * 16; }
    else             { bias = bv; orow = (ot - 4) * 16; }
    v4f bb0 = *reinterpret_cast<const v4f*>(bias + orow + kb);
    v4f bb1 = *reinterpret_cast<const v4f*>(bias + orow + kb + 4);
#pragma unroll
    for (int jj = 0; jj < 2; ++jj) {
      const int nt = half * 2 + jj;
      const int n  = n0 + nt * 16 + row_a;
      float vals[8];
#pragma unroll
      for (int r = 0; r < 8; ++r) vals[r] = acc[ii * 2 + jj][r] + (r < 4 ? bb0[r] : bb1[r - 4]);

      if (ot < 4) {
        bf16_t* dst = (ot < 2 ? qT : kT) + ((size_t)b * NN + n) * CQ + (ot & 1) * 16 + kb;
        v8bf o;
#pragma unroll
        for (int r = 0; r < 8; ++r) o[r] = (bf16_t)vals[r];
        *reinterpret_cast<v8bf*>(dst) = o;
      } else {
        const int cbase = (ot - 4) * 16 + kb;
        const int nloc  = nt * 16 + row_a;
#pragma unroll
        for (int r = 0; r < 8; ++r) vstage[(cbase + r) * 72 + nloc] = (bf16_t)vals[r];
      }
    }
  }
  __syncthreads();
  // vstage[c][n] -> vT[b][c][n0..n0+64): 128B contiguous per channel row
  {
    const int c = t;
    bf16_t* dst = vT + ((size_t)b * CC + c) * NN + n0;
    const bf16_t* srcr = vstage + c * 72;
#pragma unroll
    for (int jj = 0; jj < 4; ++jj) {
      uint4 d0 = *reinterpret_cast<const uint4*>(srcr + jj * 16);
      *reinterpret_cast<uint4*>(dst + jj * 16) = d0;
    }
  }
}

// ---------------------------------------------------------------------------
// Kernel 2: per-COLUMN softmax stats (softmax over axis n). One wave owns a
// 16-wide m-tile; k B-fragment is loop-invariant; stream 256 q A-fragments.
// shift[b][m] = max_n S[n,m] + log(sum_n exp(S[n,m]-max))
// ---------------------------------------------------------------------------
__global__ __launch_bounds__(256) void stats_kernel(
    const bf16_t* __restrict__ qT, const bf16_t* __restrict__ kT,
    float* __restrict__ shift) {
  const int t = threadIdx.x, lane = t & 31, w = t >> 5;
  const int b  = blockIdx.y;
  const int m0 = blockIdx.x * 128 + w * 16;

  const v16bf kf = load_b_rows(kT + ((size_t)b * NN + m0) * CQ, CQ, lane);
  const bf16_t* qb = qT + (size_t)b * NN * CQ;
  const v8f zero = v8f{0.f,0.f,0.f,0.f,0.f,0.f,0.f,0.f};

  float runmax = -INFINITY, runsum = 0.f;
  for (int ntile = 0; ntile < NN / 16; ++ntile) {
    v16bf af = load_a32(qb + (size_t)ntile * 16 * CQ, lane);
    v8f s = wmma_bf16(af, kf, zero);
    float fm = s[0];
#pragma unroll
    for (int r = 1; r < 8; ++r) fm = fmaxf(fm, s[r]);
    float nm  = fmaxf(runmax, fm);
    float acc = runsum * __expf(runmax - nm);
#pragma unroll
    for (int r = 0; r < 8; ++r) acc += __expf(s[r] - nm);
    runsum = acc; runmax = nm;
  }
  // merge lane L (rows 0..7 of each tile) with lane L+16 (rows 8..15)
  float om = __shfl_xor(runmax, 16, 32);
  float os = __shfl_xor(runsum, 16, 32);
  float M  = fmaxf(runmax, om);
  float Z  = runsum * __expf(runmax - M) + os * __expf(om - M);
  if (lane < 16) shift[(size_t)b * NN + m0 + lane] = M + __logf(Z);
}

// ---------------------------------------------------------------------------
// Kernel 3: out[n,c] = sum_m exp(S[n,m]-shift[m]) * v[m,c]; then
// final[b][c][n] = gamma*out + x. Block: 32 n-rows x 256 c; wave: 16n x 64c.
// shift[] preloaded to LDS via the Tensor Data Mover where available.
// ---------------------------------------------------------------------------
__global__ __launch_bounds__(256) void attn_out_kernel(
    const bf16_t* __restrict__ qT, const bf16_t* __restrict__ kT,
    const bf16_t* __restrict__ vT, const float* __restrict__ shift,
    const float* __restrict__ x, const float* __restrict__ gamma,
    float* __restrict__ out) {
  __shared__ float  shiftS[NN];                          // 16 KB
  __shared__ __align__(16) bf16_t pstage[8 * 16 * 32];   // 8 KB, wave-private P tiles
  __shared__ __align__(16) float  ostage[CC * 32];       // 32 KB

  const int t = threadIdx.x, lane = t & 31, w = t >> 5;
  const int b  = blockIdx.y;
  const int n0 = blockIdx.x * 32;
  const int wn = w >> 2, wc = w & 3;
  const int nrow0 = n0 + wn * 16;
  const int c0 = wc * 64;
  const int col = lane & 15;
  const int rhalf = (lane >> 4) * 8;

#if __has_builtin(__builtin_amdgcn_tensor_load_to_lds)
  // TDM: one wave DMAs shift[b][0..4096) (16 KB, 4B elems, 1x4096 tile) into LDS.
  if (w == 0) {
    typedef unsigned u32x4 __attribute__((ext_vector_type(4)));
    typedef int      i32x4 __attribute__((ext_vector_type(4)));
    typedef int      i32x8 __attribute__((ext_vector_type(8)));
    const unsigned lds_off = (unsigned)(size_t)(AS3 float*)shiftS;
    const unsigned long long ga =
        (unsigned long long)(size_t)(shift + (size_t)b * NN);
    u32x4 g0 = { 1u,                                   // count=1 (user D#)
                 lds_off,                              // lds_addr
                 (unsigned)ga,                         // global_addr[31:0]
                 (unsigned)((ga >> 32) & 0x01ffffffu)  // global_addr[56:32]
                   | 0x80000000u };                    // type=2 ("image")
    i32x8 g1 = { (int)0x00020000,   // data_size=2 (4 bytes)
                 (int)0x10000000,   // tensor_dim0 lo16 = 4096
                 (int)0x00010000,   // tensor_dim0 hi16=0 | tensor_dim1 lo16 = 1
                 (int)0x10000000,   // tensor_dim1 hi16=0 | tile_dim0 = 4096
                 (int)0x00000001,   // tile_dim1 = 1, tile_dim2 = 0
                 (int)4096,         // tensor_dim0_stride lo32
                 (int)0x10000000,   // stride0 hi16=0 | tensor_dim1_stride lo16 = 4096
                 0 };               // tensor_dim1_stride hi32
    i32x4 zz = {0, 0, 0, 0};
#if __has_include(<hip/amd_detail/amd_gfx1250_TDM.h>)
    i32x8 z8 = {0, 0, 0, 0, 0, 0, 0, 0};
    __builtin_amdgcn_tensor_load_to_lds(g0, g1, zz, zz, z8, 0);
#else
    __builtin_amdgcn_tensor_load_to_lds(g0, g1, zz, zz, 0);
#endif
#if __has_builtin(__builtin_amdgcn_s_wait_tensorcnt)
    __builtin_amdgcn_s_wait_tensorcnt((unsigned short)0);
#else
    asm volatile("s_wait_tensorcnt 0" ::: "memory");
#endif
  }
#else
#pragma unroll
  for (int i = 0; i < 16; ++i) shiftS[i * 256 + t] = shift[(size_t)b * NN + i * 256 + t];
#endif
  __syncthreads();

  const v16bf qf = load_a32(qT + ((size_t)b * NN + nrow0) * CQ, lane);  // loop-invariant
  const bf16_t* kbase = kT + (size_t)b * NN * CQ;
  const bf16_t* vbase = vT + (size_t)b * CC * NN;
  bf16_t* pw = pstage + w * (16 * 32);
  const v8f zero = v8f{0.f,0.f,0.f,0.f,0.f,0.f,0.f,0.f};
  v8f acc[4] = {zero, zero, zero, zero};

  for (int mi = 0; mi < NN / 32; ++mi) {
    const int m0 = mi * 32;
    v16bf kf0 = load_b_rows(kbase + (size_t)m0 * CQ, CQ, lane);
    v16bf kf1 = load_b_rows(kbase + (size_t)(m0 + 16) * CQ, CQ, lane);
    v8f s0 = wmma_bf16(qf, kf0, zero);
    v8f s1 = wmma_bf16(qf, kf1, zero);
    float sh0 = shiftS[m0 + col];
    float sh1 = shiftS[m0 + 16 + col];
    // P = exp(S - shift[m]) -> bf16, stage D-layout -> A-layout via LDS
#pragma unroll
    for (int r = 0; r < 8; ++r) {
      pw[(rhalf + r) * 32 + col]      = (bf16_t)__expf(s0[r] - sh0);
      pw[(rhalf + r) * 32 + 16 + col] = (bf16_t)__expf(s1[r] - sh1);
    }
    asm volatile("s_wait_dscnt 0" ::: "memory");  // wave is lockstep: LDS RAW safe
    v16bf pf = load_a32(pw, lane);
#pragma unroll
    for (int ct = 0; ct < 4; ++ct) {
      v16bf vf = load_b_rows(vbase + (size_t)(c0 + ct * 16) * NN + m0, NN, lane);
      acc[ct] = wmma_bf16(pf, vf, acc[ct]);
    }
  }

  // accumulators -> ostage[c][nloc] (transpose via LDS for coalesced epilogue)
#pragma unroll
  for (int ct = 0; ct < 4; ++ct) {
    float* orow = ostage + (c0 + ct * 16 + col) * 32 + wn * 16 + rhalf;
    v4f lo = {acc[ct][0], acc[ct][1], acc[ct][2], acc[ct][3]};
    v4f hi = {acc[ct][4], acc[ct][5], acc[ct][6], acc[ct][7]};
    *reinterpret_cast<v4f*>(orow)     = lo;
    *reinterpret_cast<v4f*>(orow + 4) = hi;
  }
  __syncthreads();

  const float g = gamma[0];
  {
    const int c = t;
    const float* xr   = x   + ((size_t)b * CC + c) * NN + n0;
    float*       outr = out + ((size_t)b * CC + c) * NN + n0;
    const float* osr  = ostage + c * 32;
#pragma unroll
    for (int jj = 0; jj < 8; ++jj) {
      v4f ov = *reinterpret_cast<const v4f*>(osr + jj * 4);
      v4f xv = *reinterpret_cast<const v4f*>(xr + jj * 4);
      v4f res = {g * ov[0] + xv[0], g * ov[1] + xv[1], g * ov[2] + xv[2], g * ov[3] + xv[3]};
      *reinterpret_cast<v4f*>(outr + jj * 4) = res;
    }
  }
}

// ---------------------------------------------------------------------------
extern "C" void kernel_launch(void* const* d_in, const int* in_sizes, int n_in,
                              void* d_out, int out_size, void* d_ws, size_t ws_size,
                              hipStream_t stream) {
  const float* x     = (const float*)d_in[0];
  const float* Wq    = (const float*)d_in[1];
  const float* bq    = (const float*)d_in[2];
  const float* Wk    = (const float*)d_in[3];
  const float* bk    = (const float*)d_in[4];
  const float* Wv    = (const float*)d_in[5];
  const float* bv    = (const float*)d_in[6];
  const float* gamma = (const float*)d_in[7];
  float* out = (float*)d_out;

  char* ws = (char*)d_ws;
  bf16_t* qT = (bf16_t*)(ws);                                   // 1 MB
  bf16_t* kT = (bf16_t*)(ws + (1u << 20));                      // 1 MB
  bf16_t* vT = (bf16_t*)(ws + (2u << 20));                      // 8 MB
  float* shift = (float*)(ws + (2u << 20) + (size_t)BB * CC * NN * 2);  // 64 KB

  qkv_kernel<<<dim3(NN / 64, BB), 256, 0, stream>>>(x, Wq, bq, Wk, bk, Wv, bv, qT, kT, vT);
  stats_kernel<<<dim3(NN / 128, BB), 256, 0, stream>>>(qT, kT, shift);
  attn_out_kernel<<<dim3(NN / 32, BB), 256, 0, stream>>>(qT, kT, vT, shift, x, gamma, out);
}
